// LC_FDNet_plus_44693429682846
// MI455X (gfx1250) — matrix operand order repaired and checked by reading, not compiled, not run on hardware
//
#include <hip/hip_runtime.h>
#include <hip/hip_bf16.h>

// ---------------------------------------------------------------------------
// LC_FDNet_plus forward for MI455X (gfx1250, wave32).
// All 3x3 convs with Cin in {64,128} run as implicit GEMM on
// v_wmma_f32_16x16x32_f16. Activations are f16 NHWC; conv outputs are f32
// (training-mode BN needs batch stats); BN apply re-emits f16 NHWC.
// Weights are device-packed [tap][coutPad][Cin] so the per-tap B stage is a
// contiguous b128 burst and each fragment is two 16B LDS reads.
// ---------------------------------------------------------------------------

typedef __attribute__((ext_vector_type(16))) _Float16 v16h;
typedef __attribute__((ext_vector_type(8)))  _Float16 v8h;
typedef __attribute__((ext_vector_type(8)))  float    v8f;
typedef _Float16 f16;

#define NB   8
#define HH   128
#define WW   128
#define HWF  (HH*WW)      // 16384

// ---------------------------------------------------------------------------
// Implicit-GEMM 3x3 SAME conv, f16 NHWC in, f32 out. CIN is compile-time.
// Block = 128 threads = 4 waves. Block tile: 32 pixels x 64 couts.
// Wave wv owns couts [cb*64 + wv*16, +16) and BOTH 16-pixel M-tiles
// (two accumulators sharing each B fragment). One barrier pair per tap:
// stage A (32 x CIN) + B (64 x CIN), then (CIN/32)*2 WMMAs per wave.
// ---------------------------------------------------------------------------
template <int CIN>
__global__ __launch_bounds__(128)
void k_conv3_wmma(const f16* __restrict__ in, const f16* __restrict__ wpk,
                  const float* __restrict__ bias, float* __restrict__ out,
                  int Hh, int Ww, int CoutPad, int CoutReal,
                  int biasPerBatch, int outNCHW, long wBatchStride)
{
    constexpr int CHUNKS = CIN / 32;
    const int HWl  = Hh * Ww;
    const int tile = blockIdx.x;            // 32-pixel, row-aligned tile
    const int cb   = blockIdx.y;            // 64-cout block
    const int b    = blockIdx.z;
    const int tid  = threadIdx.x;
    const int wv   = tid >> 5;
    const int lane = tid & 31;
    const int p0   = tile * 32;
    const int h    = p0 / Ww;
    const int w0   = p0 % Ww;
    const int coutBase = cb * 64 + wv * 16;

    __shared__ __attribute__((aligned(64))) f16 As[32 * CIN];
    __shared__ __attribute__((aligned(64))) f16 Bs[64 * CIN];

    const int nGlob = coutBase + (lane & 15);
    float bv = 0.f;
    if (nGlob < CoutReal)
        bv = bias[(biasPerBatch ? b * CoutReal : 0) + nGlob];
    v8f acc0 = {bv, bv, bv, bv, bv, bv, bv, bv};
    v8f acc1 = acc0;

    const f16* wb    = wpk + (long)b * wBatchStride + (long)cb * 64 * CIN;
    const f16* inRow = in + (long)b * HWl * CIN;

    // fragment source offsets (constant per lane)
    const int mRow = lane & 15;
    const int ka   = (lane < 16) ? 0 : 8;   // A: K{0..7,16..23} | K{8..15,24..31}
    const int kb   = (lane < 16) ? 0 : 8;   // B: K{0..7,16..23} | K{8..15,24..31}
    const int nLoc = wv * 16 + (lane & 15);

    for (int tap = 0; tap < 9; ++tap) {
        const int ky = tap / 3, kx = tap % 3;
        const int y  = h + ky - 1;
        const bool yok = (unsigned)y < (unsigned)Hh;
        const f16* aBase = inRow + ((long)y * Ww) * CIN;
        const f16* bBase = wb + (long)tap * CoutPad * CIN;
        if (tap < 8) // warm L2 for the next tap's weight slab (global_prefetch)
            __builtin_prefetch((const void*)(wb + (long)(tap + 1) * CoutPad * CIN), 0, 1);

        __syncthreads();
        // --- stage A: 32 pixels x CIN (zero-padded SAME), b128 per thread-step
#pragma unroll
        for (int i = 0; i < 32 * CIN / 8 / 128; ++i) {
            const int e = i * 128 + tid;          // 8-f16 granule index
            const int m = e / (CIN / 8);
            const int g = e % (CIN / 8);
            const int x = w0 + m + kx - 1;
            v8h v = {};
            if (yok && (unsigned)x < (unsigned)Ww)
                v = *(const v8h*)&aBase[(long)x * CIN + g * 8];
            *(v8h*)&As[m * CIN + g * 8] = v;
        }
        // --- stage B: 64 couts x CIN, fully contiguous in packed layout
#pragma unroll
        for (int i = 0; i < 64 * CIN / 8 / 128; ++i) {
            const int e = i * 128 + tid;
            *(v8h*)&Bs[e * 8] = *(const v8h*)&bBase[e * 8];
        }
        __syncthreads();

        // --- (CIN/32) chunks x 2 M-tiles of WMMA, B fragment reused across M
#pragma unroll
        for (int ch = 0; ch < CHUNKS; ++ch) {
            union { v16h v; v8h h2[2]; } bf;
            bf.h2[0] = *(const v8h*)&Bs[nLoc * CIN + ch * 32 + kb];
            bf.h2[1] = *(const v8h*)&Bs[nLoc * CIN + ch * 32 + kb + 16];
            union { v16h v; v8h h2[2]; } af0, af1;
            af0.h2[0] = *(const v8h*)&As[mRow * CIN + ch * 32 + ka];
            af0.h2[1] = *(const v8h*)&As[mRow * CIN + ch * 32 + ka + 16];
            af1.h2[0] = *(const v8h*)&As[(16 + mRow) * CIN + ch * 32 + ka];
            af1.h2[1] = *(const v8h*)&As[(16 + mRow) * CIN + ch * 32 + ka + 16];
            acc0 = __builtin_amdgcn_wmma_f32_16x16x32_f16(
                false, af0.v, false, bf.v, (short)0, acc0, false, false);
            acc1 = __builtin_amdgcn_wmma_f32_16x16x32_f16(
                false, af1.v, false, bf.v, (short)0, acc1, false, false);
        }
    }

    if (nGlob < CoutReal) {
        const int mOfs = (lane >= 16) ? 8 : 0;
#pragma unroll
        for (int r = 0; r < 8; ++r) {
            const int m0 = r + mOfs;
            const float v0 = acc0[r];
            const float v1 = acc1[r];
            if (outNCHW) {
                float* o = out + (((long)b * CoutReal + nGlob) * HWl) + p0;
                o[m0]      = v0;
                o[m0 + 16] = v1;
            } else {
                out[(((long)b * HWl + p0 + m0) * CoutReal) + nGlob]      = v0;
                out[(((long)b * HWl + p0 + m0 + 16) * CoutReal) + nGlob] = v1;
            }
        }
    }
}

// Pack f32 OIHW [CoutReal][Cin][3][3] -> f16 [tap][CoutPad][Cin]
__global__ void k_weight_pack(const float* __restrict__ w, f16* __restrict__ wpk,
                              int Cin, int CoutPad, int CoutReal)
{
    const int total = 9 * CoutPad * Cin;
    int idx = blockIdx.x * blockDim.x + threadIdx.x;
    if (idx >= total) return;
    const int ci  = idx % Cin;
    const int n   = (idx / Cin) % CoutPad;
    const int tap = idx / (Cin * CoutPad);
    float v = 0.f;
    if (n < CoutReal) v = w[((long)n * Cin + ci) * 9 + tap];
    wpk[idx] = (f16)v;
}

// Dynamic kernels: k = temp @ lcw + lcb packed [b][tap][cout(64)][cin(64)].
__global__ void k_dyn_pack(const float* __restrict__ temp, const float* __restrict__ lcw,
                           const float* __restrict__ lcb, f16* __restrict__ wpk)
{
    int idx = blockIdx.x * blockDim.x + threadIdx.x;
    if (idx >= NB * 9 * 64 * 64) return;
    const int ci  = idx & 63;
    const int n   = (idx >> 6) & 63;
    const int tap = (idx >> 12) % 9;
    const int b   = (idx >> 12) / 9;
    const int col = (n * 64 + ci) * 9 + tap;
    float s = lcb[col];
    for (int j = 0; j < 64; ++j) s += temp[b * 64 + j] * lcw[(long)j * 36864 + col];
    wpk[idx] = (f16)s;
}

// Tiny GEMM: out[M,N] = A[M,K] @ W[K,N] + bias[N]
__global__ void k_linear(const float* __restrict__ A, const float* __restrict__ Wm,
                         const float* __restrict__ bias, float* __restrict__ out,
                         int M, int K, int N)
{
    int i = blockIdx.x * blockDim.x + threadIdx.x;
    if (i >= M * N) return;
    const int m = i / N, n = i % N;
    float s = bias[n];
    for (int k = 0; k < K; ++k) s += A[m * K + k] * Wm[k * N + n];
    out[i] = s;
}

// First conv: f32 NCHW [B,4,H,W] -> f32 NHWC [B,HW,64] (cheap, direct)
__global__ void k_conv_in4(const float* __restrict__ in, const float* __restrict__ w,
                           const float* __restrict__ bias, float* __restrict__ out)
{
    int i = blockIdx.x * blockDim.x + threadIdx.x;
    if (i >= NB * HWF * 64) return;
    const int co = i & 63;
    const int p  = (i >> 6) & (HWF - 1);
    const int b  = i >> 20;
    const int h = p >> 7, x0 = p & 127;
    float s = bias[co];
    for (int ci = 0; ci < 4; ++ci)
        for (int ky = 0; ky < 3; ++ky) {
            const int y = h + ky - 1;
            if ((unsigned)y >= (unsigned)HH) continue;
            for (int kx = 0; kx < 3; ++kx) {
                const int x = x0 + kx - 1;
                if ((unsigned)x >= (unsigned)WW) continue;
                s += in[(((long)b * 4 + ci) * HH + y) * WW + x]
                   * w[((co * 4 + ci) * 3 + ky) * 3 + kx];
            }
        }
    out[i] = s;
}

// Cout=1 head: f16 NHWC (C=64) -> f32, optional relu
__global__ void k_conv_co1(const f16* __restrict__ in, const float* __restrict__ w,
                           const float* __restrict__ bias, int relu,
                           float* __restrict__ out, int Hh, int Ww)
{
    const int HWl = Hh * Ww;
    int i = blockIdx.x * blockDim.x + threadIdx.x;
    if (i >= NB * HWl) return;
    const int p = i % HWl, b = i / HWl;
    const int h = p / Ww, x0 = p % Ww;
    float s = bias[0];
    for (int ky = 0; ky < 3; ++ky) {
        const int y = h + ky - 1;
        if ((unsigned)y >= (unsigned)Hh) continue;
        for (int kx = 0; kx < 3; ++kx) {
            const int x = x0 + kx - 1;
            if ((unsigned)x >= (unsigned)Ww) continue;
            const f16* src = in + ((long)b * HWl + (long)y * Ww + x) * 64;
            const int tap = ky * 3 + kx;
#pragma unroll
            for (int ci = 0; ci < 64; ++ci) s += (float)src[ci] * w[ci * 9 + tap];
        }
    }
    if (relu) s = fmaxf(s, 0.f);
    out[i] = s;
}

__global__ void k_zero(float* p, int n)
{
    int i = blockIdx.x * blockDim.x + threadIdx.x;
    if (i < n) p[i] = 0.f;
}

// Per-channel sum / sumsq over NHWC rows via f32 atomics (accum[2C])
__global__ void k_bn_stats(const float* __restrict__ x, float* __restrict__ accum,
                           int rows, int C)
{
    const int tid = threadIdx.x;
    const int c   = tid % C;
    const int rpb = blockDim.x / C;
    int r = blockIdx.x * rpb + tid / C;
    const int stride = gridDim.x * rpb;
    float s = 0.f, s2 = 0.f;
    for (; r < rows; r += stride) {
        const float v = x[(long)r * C + c];
        s += v; s2 += v * v;
    }
    atomicAdd(&accum[c], s);
    atomicAdd(&accum[C + c], s2);
}

// y = g*(x-m)*rsqrt(v+eps)+b (+residual)(relu) -> f16 NHWC
__global__ void k_bn_apply(const float* __restrict__ x, const float* __restrict__ accum,
                           const float* __restrict__ g, const float* __restrict__ bt,
                           const f16* __restrict__ res, f16* __restrict__ out,
                           long rows, int C, float invCount, int relu)
{
    long i = (long)blockIdx.x * blockDim.x + threadIdx.x;
    if (i >= rows * C) return;
    const int c = (int)(i % C);
    const float m   = accum[c] * invCount;
    const float var = accum[C + c] * invCount - m * m;
    float y = g[c] * (x[i] - m) * rsqrtf(var + 1e-5f) + bt[c];
    if (res)  y += (float)res[i];
    if (relu) y = fmaxf(y, 0.f);
    out[i] = (f16)y;
}

__global__ void k_maxpool2(const f16* __restrict__ in, f16* __restrict__ out,
                           int Hi, int Wi)
{
    const int Ho = Hi >> 1, Wo = Wi >> 1;
    int i = blockIdx.x * blockDim.x + threadIdx.x;
    if (i >= NB * Ho * Wo * 64) return;
    const int c  = i & 63;
    const int wo = (i >> 6) % Wo;
    const int ho = ((i >> 6) / Wo) % Ho;
    const int b  = (i >> 6) / (Wo * Ho);
    const f16* p00 = in + (((long)b * Hi + 2 * ho) * Wi + 2 * wo) * 64 + c;
    const f16* p10 = p00 + (long)Wi * 64;
    const float a = (float)p00[0], bb = (float)p00[64];
    const float cc = (float)p10[0], dd = (float)p10[64];
    out[i] = (f16)fmaxf(fmaxf(a, bb), fmaxf(cc, dd));
}

__global__ void k_f32_to_f16(const float* __restrict__ x, f16* __restrict__ y, long n)
{
    long i = (long)blockIdx.x * blockDim.x + threadIdx.x;
    if (i < n) y[i] = (f16)x[i];
}

// vt tail: BN(C=1)+relu per element, per-sample mean, th = 10*mean
__global__ void k_bn1_mean(const float* __restrict__ x, const float* __restrict__ accum,
                           const float* __restrict__ g, const float* __restrict__ bt,
                           float* __restrict__ th, int per, float invCount)
{
    __shared__ float red[256];
    const int b = blockIdx.x, tid = threadIdx.x;
    const float m   = accum[0] * invCount;
    const float var = accum[1] * invCount - m * m;
    const float gg = g[0], bb = bt[0];
    float s = 0.f;
    for (int p = tid; p < per; p += 256) {
        const float y = gg * (x[b * per + p] - m) * rsqrtf(var + 1e-5f) + bb;
        s += fmaxf(y, 0.f);
    }
    red[tid] = s;
    __syncthreads();
    for (int o = 128; o > 0; o >>= 1) {
        if (tid < o) red[tid] += red[tid + o];
        __syncthreads();
    }
    if (tid == 0) th[b] = 10.f * red[0] / (float)per;
}

__global__ void k_finalize(const float* __restrict__ res, const float* __restrict__ ref,
                           const float* __restrict__ gt, const float* __restrict__ ev,
                           const float* __restrict__ th, float* __restrict__ pred,
                           float* __restrict__ qres, float* __restrict__ mask)
{
    int i = blockIdx.x * blockDim.x + threadIdx.x;
    if (i >= NB * HWF) return;
    const int b = i >> 14;
    float p = res[i] + ref[i];
    p = fminf(fmaxf(p, 0.f), 255.f);
    pred[i] = p;
    qres[i] = rintf(p - gt[i]) + 255.f;           // ste_round -> int32 +255 (as f32)
    mask[i] = (ev[i] < th[b]) ? 1.f : 0.f;
}

// ---------------------------------------------------------------------------

static inline int cdiv(long a, long b) { return (int)((a + b - 1) / b); }

extern "C" void kernel_launch(void* const* d_in, const int* in_sizes, int n_in,
                              void* d_out, int out_size, void* d_ws, size_t ws_size,
                              hipStream_t stream)
{
    (void)in_sizes; (void)n_in; (void)out_size; (void)ws_size;
    const float* x_in = (const float*)d_in[0];
    const float* qf   = (const float*)d_in[1];
    const float* gt   = (const float*)d_in[2];
    const float* ref  = (const float*)d_in[3];
    const float* P[58];
    for (int i = 0; i < 58; ++i) P[i] = (const float*)d_in[4 + i];
    // P map: 0..3 enc(c1w,c1b,g1,bb1); 4..11 rb1; 12..19 rb2; 20..27 rb3;
    // 28..35 l1w,l1b,l2w,l2b,lcw,lcb,lbw,lbb; 36..41 pmf(c1w,c1b,g,b,c2w,c2b);
    // 42..45 res_w,res_b,ev_w,ev_b; 46..57 vt(c1w,c1b,g1,b1,c2w,c2b,g2,b2,c3w,c3b,g3,b3)

    float* outp = (float*)d_out;
    const long o_pred = 0, o_qres = 131072, o_ev = 262144, o_th = 393216,
               o_mask = 393224, o_pmf = 524296;

    // ---- workspace bump allocation ----
    char* ws = (char*)d_ws;
    size_t off = 0;
    auto alloc = [&](size_t bytes) -> void* {
        void* p = ws + off;
        off = (off + bytes + 255) & ~(size_t)255;
        return p;
    };
    const size_t ACT16 = (size_t)NB * HWF * 64 * 2;
    f16*  x0    = (f16*)alloc(ACT16);
    f16*  x1    = (f16*)alloc(ACT16);
    f16*  x2    = (f16*)alloc(ACT16);
    f16*  pmfh  = (f16*)alloc((size_t)NB * HWF * 128 * 2);
    f16*  vhalfA= (f16*)alloc((size_t)NB * 64 * 64 * 64 * 2);
    f16*  vhalfB= (f16*)alloc((size_t)NB * 64 * 64 * 64 * 2);
    f16*  vq    = (f16*)alloc((size_t)NB * 32 * 32 * 64 * 2);
    float* cf32 = (float*)alloc((size_t)NB * HWF * 128 * 4);
    f16*  wpk   = (f16*)alloc((size_t)9 * 512 * 128 * 2);      // max pack (pmf_c2)
    f16*  wpkD  = (f16*)alloc((size_t)NB * 9 * 64 * 64 * 2);
    float* resf = (float*)alloc((size_t)NB * HWF * 4);
    float* vt3f = (float*)alloc((size_t)NB * 1024 * 4);
    float* stats= (float*)alloc(4096);
    float* t1f  = (float*)alloc(2048);
    float* tmpf = (float*)alloc(2048);
    float* dynb = (float*)alloc(2048);

    const long rowsF = (long)NB * HWF;

    // ===== encoder first conv (Cin=4, direct) + BN + relu =====
    k_conv_in4<<<cdiv(rowsF * 64, 256), 256, 0, stream>>>(x_in, P[0], P[1], cf32);
    k_zero<<<1, 256, 0, stream>>>(stats, 256);
    k_bn_stats<<<256, 256, 0, stream>>>(cf32, stats, (int)rowsF, 64);
    k_bn_apply<<<cdiv(rowsF * 64, 256), 256, 0, stream>>>(
        cf32, stats, P[2], P[3], (const f16*)nullptr, x0, rowsF, 64, 1.f / rowsF, 1);

    // ===== 3 residual blocks (WMMA convs) =====
    f16* bufs[3] = {x0, x1, x2};
    int curi = 0;
    const dim3 gFull(HWF / 32, 1, NB);
    for (int rb = 0; rb < 3; ++rb) {
        const float** rp = P + 4 + rb * 8;
        f16* cur  = bufs[curi];
        f16* mid  = bufs[(curi + 1) % 3];
        f16* outb = bufs[(curi + 2) % 3];
        k_weight_pack<<<cdiv(9 * 64 * 64, 256), 256, 0, stream>>>(rp[0], wpk, 64, 64, 64);
        k_conv3_wmma<64><<<gFull, 128, 0, stream>>>(cur, wpk, rp[1], cf32, HH, WW, 64, 64, 0, 0, 0);
        k_zero<<<1, 256, 0, stream>>>(stats, 256);
        k_bn_stats<<<256, 256, 0, stream>>>(cf32, stats, (int)rowsF, 64);
        k_bn_apply<<<cdiv(rowsF * 64, 256), 256, 0, stream>>>(
            cf32, stats, rp[2], rp[3], (const f16*)nullptr, mid, rowsF, 64, 1.f / rowsF, 1);
        k_weight_pack<<<cdiv(9 * 64 * 64, 256), 256, 0, stream>>>(rp[4], wpk, 64, 64, 64);
        k_conv3_wmma<64><<<gFull, 128, 0, stream>>>(mid, wpk, rp[5], cf32, HH, WW, 64, 64, 0, 0, 0);
        k_zero<<<1, 256, 0, stream>>>(stats, 256);
        k_bn_stats<<<256, 256, 0, stream>>>(cf32, stats, (int)rowsF, 64);
        k_bn_apply<<<cdiv(rowsF * 64, 256), 256, 0, stream>>>(
            cf32, stats, rp[6], rp[7], cur, outb, rowsF, 64, 1.f / rowsF, 1);
        curi = (curi + 2) % 3;
    }
    f16* feat = bufs[curi];

    // ===== qf -> temp -> dynamic kernels / bias =====
    k_linear<<<2, 256, 0, stream>>>(qf, P[28], P[29], t1f, NB, 4, 64);
    k_linear<<<2, 256, 0, stream>>>(t1f, P[30], P[31], tmpf, NB, 64, 64);
    k_linear<<<2, 256, 0, stream>>>(tmpf, P[34], P[35], dynb, NB, 64, 64);
    k_dyn_pack<<<cdiv(NB * 9 * 64 * 64, 256), 256, 0, stream>>>(tmpf, P[32], P[33], wpkD);

    // dynamic grouped conv (per-sample weights + per-sample bias) -> f
    k_conv3_wmma<64><<<gFull, 128, 0, stream>>>(feat, wpkD, dynb, cf32, HH, WW, 64, 64,
                                                1, 0, (long)9 * 64 * 64);
    f16* fh = bufs[(curi + 1) % 3];
    k_f32_to_f16<<<cdiv(rowsF * 64, 256), 256, 0, stream>>>(cf32, fh, rowsF * 64);

    // ===== heads =====
    k_conv_co1<<<cdiv(rowsF, 256), 256, 0, stream>>>(fh, P[42], P[43], 0, resf, HH, WW);
    k_conv_co1<<<cdiv(rowsF, 256), 256, 0, stream>>>(fh, P[44], P[45], 1, outp + o_ev, HH, WW);

    // pmf: 64->128 (WMMA) + BN + relu, then 128->511 (WMMA, padded to 512, NCHW out)
    k_weight_pack<<<cdiv(9 * 128 * 64, 256), 256, 0, stream>>>(P[36], wpk, 64, 128, 128);
    k_conv3_wmma<64><<<dim3(HWF / 32, 2, NB), 128, 0, stream>>>(fh, wpk, P[37], cf32,
                                                                HH, WW, 128, 128, 0, 0, 0);
    k_zero<<<2, 256, 0, stream>>>(stats, 512);
    k_bn_stats<<<256, 256, 0, stream>>>(cf32, stats, (int)rowsF, 128);
    k_bn_apply<<<cdiv(rowsF * 128, 256), 256, 0, stream>>>(
        cf32, stats, P[38], P[39], (const f16*)nullptr, pmfh, rowsF, 128, 1.f / rowsF, 1);
    k_weight_pack<<<cdiv(9 * 512 * 128, 256), 256, 0, stream>>>(P[40], wpk, 128, 512, 511);
    k_conv3_wmma<128><<<dim3(HWF / 32, 8, NB), 128, 0, stream>>>(pmfh, wpk, P[41],
                                                                 outp + o_pmf,
                                                                 HH, WW, 512, 511, 0, 1, 0);

    // ===== variance-threshold path =====
    k_weight_pack<<<cdiv(9 * 64 * 64, 256), 256, 0, stream>>>(P[46], wpk, 64, 64, 64);
    k_conv3_wmma<64><<<gFull, 128, 0, stream>>>(fh, wpk, P[47], cf32, HH, WW, 64, 64, 0, 0, 0);
    k_zero<<<1, 256, 0, stream>>>(stats, 256);
    k_bn_stats<<<256, 256, 0, stream>>>(cf32, stats, (int)rowsF, 64);
    f16* vt1 = bufs[(curi + 2) % 3];
    k_bn_apply<<<cdiv(rowsF * 64, 256), 256, 0, stream>>>(
        cf32, stats, P[48], P[49], (const f16*)nullptr, vt1, rowsF, 64, 1.f / rowsF, 1);
    k_maxpool2<<<cdiv((long)NB * 64 * 64 * 64, 256), 256, 0, stream>>>(vt1, vhalfA, HH, WW);

    const long rowsH = (long)NB * 64 * 64;
    k_weight_pack<<<cdiv(9 * 64 * 64, 256), 256, 0, stream>>>(P[50], wpk, 64, 64, 64);
    k_conv3_wmma<64><<<dim3(64 * 64 / 32, 1, NB), 128, 0, stream>>>(vhalfA, wpk, P[51], cf32,
                                                                    64, 64, 64, 64, 0, 0, 0);
    k_zero<<<1, 256, 0, stream>>>(stats, 256);
    k_bn_stats<<<128, 256, 0, stream>>>(cf32, stats, (int)rowsH, 64);
    k_bn_apply<<<cdiv(rowsH * 64, 256), 256, 0, stream>>>(
        cf32, stats, P[52], P[53], (const f16*)nullptr, vhalfB, rowsH, 64, 1.f / rowsH, 1);
    k_maxpool2<<<cdiv((long)NB * 32 * 32 * 64, 256), 256, 0, stream>>>(vhalfB, vq, 64, 64);

    k_conv_co1<<<cdiv((long)NB * 1024, 256), 256, 0, stream>>>(vq, P[54], P[55], 0, vt3f, 32, 32);
    k_zero<<<1, 256, 0, stream>>>(stats, 8);
    k_bn_stats<<<8, 256, 0, stream>>>(vt3f, stats, NB * 1024, 1);
    k_bn1_mean<<<NB, 256, 0, stream>>>(vt3f, stats, P[56], P[57], outp + o_th,
                                       1024, 1.f / (NB * 1024));

    // ===== outputs: pred / q_res / mask =====
    k_finalize<<<cdiv(rowsF, 256), 256, 0, stream>>>(resf, ref, gt, outp + o_ev,
                                                     outp + o_th, outp + o_pred,
                                                     outp + o_qres, outp + o_mask);
}
